// MultiHeadAttention_18442589569044
// MI455X (gfx1250) — matrix operand order repaired and checked by reading, not compiled, not run on hardware
//
#include <hip/hip_runtime.h>
#include <hip/hip_bf16.h>
#include <math.h>

typedef _Float16 v16h __attribute__((ext_vector_type(16)));
typedef _Float16 v8h  __attribute__((ext_vector_type(8)));
typedef _Float16 v4h  __attribute__((ext_vector_type(4)));
typedef float    v8f  __attribute__((ext_vector_type(8)));

#define D_MODEL 1024
#define SEQ     2048
#define BATCH   2
#define HEADS   16
#define DK      64
#define TOKENS  (BATCH*SEQ)   // 4096

static __device__ __forceinline__ v8f wmma16(v16h a, v16h b, v8f c) {
  return __builtin_amdgcn_wmma_f32_16x16x32_f16(false, a, false, b, (short)0, c, false, false);
}

static __device__ __forceinline__ v4h cvt4(float4 x) {
  v4h h = { (_Float16)x.x, (_Float16)x.y, (_Float16)x.z, (_Float16)x.w };
  return h;
}

// nibble-select constants for a xor-butterfly within 16-lane rows
constexpr unsigned selLo(int d) {
  unsigned x = 0;
  for (int j = 0; j < 8; ++j) x |= ((unsigned)((j ^ d) & 15)) << (4 * j);
  return x;
}
constexpr unsigned selHi(int d) {
  unsigned x = 0;
  for (int j = 8; j < 16; ++j) x |= ((unsigned)((j ^ d) & 15)) << (4 * (j - 8));
  return x;
}
static __device__ __forceinline__ float permxor16(float v, unsigned lo, unsigned hi) {
  unsigned u = __float_as_uint(v);
  unsigned r = __builtin_amdgcn_permlane16(u, u, lo, hi, false, false);
  return __uint_as_float(r);
}

// -----------------------------------------------------------------------------
// out[m,n] = sum_k X[m,k] * W[n,k] + bias[n]      (M=4096, N=1024, K=1024)
// WG: 256 threads = 8 waves; WG tile 128(M) x 128(N); wave tile 32(M) x 64(N).
// Double-buffered LDS in WMMA fragment order; 8 WMMA / wave / k-step.
// MODE 0: f16 out [B,H,S,DK] (Q,K); 1: f16 [B,H,DK,S] (V^T); 2: f32 [tok,D].
// -----------------------------------------------------------------------------
template<int MODE>
__global__ __launch_bounds__(256)
void proj_gemm(const float* __restrict__ X, const float* __restrict__ W,
               const float* __restrict__ bias, void* __restrict__ out)
{
  __shared__ alignas(64) _Float16 Af[2][8][32][16];
  __shared__ alignas(64) _Float16 Bf[2][8][32][16];
  const int tid  = threadIdx.x;
  const int lane = tid & 31;
  const int w    = tid >> 5;
  const int wm   = w & 3, wn = w >> 2;       // 4 x 2 wave grid
  const int m0   = blockIdx.x * 128;
  const int n0   = blockIdx.y * 128;

  // per-thread staging coordinates (4 float4 chunks each for A and B)
  int srow[4], sk4[4];
  #pragma unroll
  for (int j = 0; j < 4; ++j) {
    int i = tid + j * 256;
    srow[j] = i >> 3;
    sk4[j]  = (i & 7) << 2;
  }

  float4 ra[4], rb[4];
  v8f acc[2][4] = {};

  auto gload = [&](int ks) {
    const int k0 = ks * 32;
    #pragma unroll
    for (int j = 0; j < 4; ++j) {
      ra[j] = *(const float4*)(X + (size_t)(m0 + srow[j]) * D_MODEL + k0 + sk4[j]);
      rb[j] = *(const float4*)(W + (size_t)(n0 + srow[j]) * D_MODEL + k0 + sk4[j]);
    }
  };
  auto gstore = [&](int p) {
    #pragma unroll
    for (int j = 0; j < 4; ++j) {
      int row = srow[j], k4 = sk4[j];
      int sla = (row & 15) | (((k4 & 15) >= 8) ? 16 : 0);
      int ea  = (k4 & 7) + ((k4 >= 16) ? 8 : 0);
      *(v4h*)&Af[p][row >> 4][sla][ea] = cvt4(ra[j]);
      int slb = (row & 15) | (((k4 & 15) >= 8) ? 16 : 0);
      int eb  = (k4 & 7) + ((k4 >= 16) ? 8 : 0);
      *(v4h*)&Bf[p][row >> 4][slb][eb] = cvt4(rb[j]);
    }
  };
  auto gcompute = [&](int p) {
    v16h a0 = *(const v16h*)&Af[p][2*wm + 0][lane][0];
    v16h a1 = *(const v16h*)&Af[p][2*wm + 1][lane][0];
    v16h b0 = *(const v16h*)&Bf[p][4*wn + 0][lane][0];
    v16h b1 = *(const v16h*)&Bf[p][4*wn + 1][lane][0];
    v16h b2 = *(const v16h*)&Bf[p][4*wn + 2][lane][0];
    v16h b3 = *(const v16h*)&Bf[p][4*wn + 3][lane][0];
    acc[0][0] = wmma16(a0, b0, acc[0][0]);
    acc[0][1] = wmma16(a0, b1, acc[0][1]);
    acc[0][2] = wmma16(a0, b2, acc[0][2]);
    acc[0][3] = wmma16(a0, b3, acc[0][3]);
    acc[1][0] = wmma16(a1, b0, acc[1][0]);
    acc[1][1] = wmma16(a1, b1, acc[1][1]);
    acc[1][2] = wmma16(a1, b2, acc[1][2]);
    acc[1][3] = wmma16(a1, b3, acc[1][3]);
  };

  gload(0); gstore(0);
  __syncthreads();
  for (int ks = 0; ks < 32; ++ks) {
    if (ks + 1 < 32) gload(ks + 1);
    gcompute(ks & 1);
    if (ks + 1 < 32) gstore((ks + 1) & 1);
    __syncthreads();
  }

  // epilogue: bias add + scatter (compile-time MODE -> branch-free)
  const int hi = lane >> 4, ln = lane & 15;
  #pragma unroll
  for (int i = 0; i < 2; ++i) {
    #pragma unroll
    for (int j = 0; j < 4; ++j) {
      int ncol = n0 + (4*wn + j) * 16 + ln;
      float bj = bias[ncol];
      int hh = ncol >> 6, d = ncol & 63;
      #pragma unroll
      for (int r = 0; r < 8; ++r) {
        int mrow = m0 + (2*wm + i) * 16 + r + 8*hi;
        float val = acc[i][j][r] + bj;
        int b = mrow >> 11, s = mrow & (SEQ - 1);
        if (MODE == 0) {
          ((_Float16*)out)[(((size_t)(b*HEADS + hh))*SEQ + s)*DK + d] = (_Float16)val;
        } else if (MODE == 1) {
          ((_Float16*)out)[(((size_t)(b*HEADS + hh))*DK + d)*SEQ + s] = (_Float16)val;
        } else {
          ((float*)out)[(size_t)mrow * D_MODEL + ncol] = val;
        }
      }
    }
  }
}

// -----------------------------------------------------------------------------
// Flash attention. Block = 256 threads = 8 waves, all on the same (b,h).
// Each wave owns 16 query rows; block covers 128 rows. K/V tiles
// (32 keys x 64 d, f16) double-buffered in LDS, shared by all 8 waves.
// Row-max reduced with v_permlane16 (VALU); row-sum via WMMA against a
// ones matrix (consistent with the f16 P used in the P@V numerator).
// 1/sqrt(dk) folded into the Q fragment once at load.
// Q/K f16 [B,H,S,64]; V f16 transposed [B,H,64,S]; ctx f32 [tokens, 1024].
// -----------------------------------------------------------------------------
__global__ __launch_bounds__(256)
void flash_attn(const _Float16* __restrict__ Qh, const _Float16* __restrict__ Kh,
                const _Float16* __restrict__ Vt, const int* __restrict__ mask,
                float* __restrict__ ctx)
{
  __shared__ alignas(64) _Float16 Kt[2][32][64];     // [buf][key][d]     8 KB
  __shared__ alignas(64) _Float16 Vl[2][64][32];     // [buf][d][key]     8 KB
  __shared__ alignas(64) _Float16 Plds[8][16][32];   // [wave]            8 KB

  const int tid  = threadIdx.x;
  const int lane = tid & 31;
  const int wid  = tid >> 5;                 // 0..7
  const int bh   = blockIdx.x >> 4;          // 0..31
  const int qblk = blockIdx.x & 15;          // 16 blocks of 128 rows
  const int b    = bh >> 4, h = bh & 15;
  const int q0   = qblk * 128 + wid * 16;    // this wave's 16 query rows
  const int hi   = lane >> 4, ln = lane & 15;

  const size_t qkbase = (size_t)bh * SEQ * DK;
  const size_t vbase  = (size_t)bh * DK * SEQ;
  const int*   km     = mask + b * SEQ;

  // Q A-fragment set: 16 rows x 64 d (two K=32 steps), loaded in A layout,
  // pre-scaled by 1/sqrt(dk) so scores come out of the WMMA already scaled.
  const _Float16 qscale = (_Float16)0.125f;
  v16h aq[2];
  {
    const _Float16* qrow = Qh + qkbase + (size_t)(q0 + ln) * DK;
    #pragma unroll
    for (int kk = 0; kk < 2; ++kk) {
      v8h lo = *(const v8h*)(qrow + kk*32 + hi*16);
      v8h hv = *(const v8h*)(qrow + kk*32 + hi*16 + 8);
      v16h a;
      #pragma unroll
      for (int t = 0; t < 8; ++t) { a[t] = lo[t] * qscale; a[t+8] = hv[t] * qscale; }
      aq[kk] = a;
    }
  }

  // all-ones B fragment: rowsum(P) = P x ones via one WMMA
  v16h vones;
  #pragma unroll
  for (int t = 0; t < 16; ++t) vones[t] = (_Float16)1.0f;

  float mrow[8], lrow[8];
  v8f acc[4] = {};
  #pragma unroll
  for (int r = 0; r < 8; ++r) { mrow[r] = -INFINITY; lrow[r] = 0.f; }

  // cooperative K/V tile staging: 256 threads, 1 uint4 (8 f16) chunk each
  const int krow = tid >> 3, koff = (tid & 7) * 8;   // K tile: 32 x 64
  const int vrow = tid >> 2, voff = (tid & 3) * 8;   // V tile: 64 x 32
  uint4 rk, rv;
  auto sload = [&](int it) {
    const int kb = it * 32;
    rk = *(const uint4*)(Kh + qkbase + (size_t)(kb + krow) * DK + koff);
    rv = *(const uint4*)(Vt + vbase + (size_t)vrow * SEQ + kb + voff);
  };
  auto sstore = [&](int p) {
    *(uint4*)&Kt[p][krow][koff] = rk;
    *(uint4*)&Vl[p][vrow][voff] = rv;
  };

  auto compute = [&](int p, int kb) {
    const int mk0 = km[kb + ln];
    const int mk1 = km[kb + 16 + ln];

    // ---- scores: S = (Q/8) K^T for 32 keys (two 16x16 C frags, K-dim 64)
    v8f c0 = {}, c1 = {};
    {
      v16h t0 = *(const v16h*)&Kt[p][ln][hi*16];
      v16h t1 = *(const v16h*)&Kt[p][ln][32 + hi*16];
      c0 = wmma16(aq[0], t0, c0);
      c0 = wmma16(aq[1], t1, c0);
    }
    {
      v16h t0 = *(const v16h*)&Kt[p][16 + ln][hi*16];
      v16h t1 = *(const v16h*)&Kt[p][16 + ln][32 + hi*16];
      c1 = wmma16(aq[0], t0, c1);
      c1 = wmma16(aq[1], t1, c1);
    }

    float s0[8], s1[8], mx[8];
    #pragma unroll
    for (int r = 0; r < 8; ++r) {
      s0[r] = mk0 ? c0[r] : -1.0e9f;
      s1[r] = mk1 ? c1[r] : -1.0e9f;
      mx[r] = fmaxf(s0[r], s1[r]);
    }
    // row max across the 16-lane row group: VALU butterfly (v_permlane16)
    #pragma unroll
    for (int st = 0; st < 4; ++st) {
      const int d = 1 << st;
      const unsigned lo = selLo(d), hiw = selHi(d);
      #pragma unroll
      for (int r = 0; r < 8; ++r)
        mx[r] = fmaxf(mx[r], permxor16(mx[r], lo, hiw));
    }
    float alpha[8];
    #pragma unroll
    for (int r = 0; r < 8; ++r) {
      float mn = fmaxf(mrow[r], mx[r]);
      alpha[r] = __expf(mrow[r] - mn);
      s0[r] = __expf(s0[r] - mn);
      s1[r] = __expf(s1[r] - mn);
      mrow[r] = mn;
    }
    #pragma unroll
    for (int nf = 0; nf < 4; ++nf)
      #pragma unroll
      for (int r = 0; r < 8; ++r)
        acc[nf][r] *= alpha[r];

    // P: C-layout -> A-layout via wave-private LDS (in-order DS per wave)
    #pragma unroll
    for (int r = 0; r < 8; ++r) {
      Plds[wid][r + 8*hi][ln]      = (_Float16)s0[r];
      Plds[wid][r + 8*hi][16 + ln] = (_Float16)s1[r];
    }
    v16h ap;
    {
      v8h lo = *(const v8h*)&Plds[wid][ln][hi*16];
      v8h hv = *(const v8h*)&Plds[wid][ln][hi*16 + 8];
      #pragma unroll
      for (int t = 0; t < 8; ++t) { ap[t] = lo[t]; ap[t+8] = hv[t]; }
    }
    // ---- row sums of f16 P via WMMA against ones (all columns = rowsum)
    {
      v8f z = {};
      v8f rsf = wmma16(ap, vones, z);
      #pragma unroll
      for (int r = 0; r < 8; ++r)
        lrow[r] = lrow[r] * alpha[r] + rsf[r];
    }
    // ---- O += P @ V  (V fragment loaded just-in-time per WMMA)
    #pragma unroll
    for (int nf = 0; nf < 4; ++nf) {
      v16h bv = *(const v16h*)&Vl[p][nf*16 + ln][hi*16];
      acc[nf] = wmma16(ap, bv, acc[nf]);
    }
  };

  sload(0); sstore(0);
  __syncthreads();
  for (int it = 0; it < SEQ / 32; ++it) {
    if (it + 1 < SEQ / 32) sload(it + 1);
    compute(it & 1, it * 32);
    if (it + 1 < SEQ / 32) sstore((it + 1) & 1);
    __syncthreads();
  }

  // normalize and write context (token-major fp32 for the output projection)
  #pragma unroll
  for (int r = 0; r < 8; ++r) {
    float inv = 1.0f / lrow[r];
    int srow = q0 + r + 8*hi;
    size_t orow = ((size_t)(b*SEQ + srow)) * D_MODEL + h*DK + ln;
    #pragma unroll
    for (int nf = 0; nf < 4; ++nf)
      ctx[orow + nf*16] = acc[nf][r] * inv;
  }
}

extern "C" void kernel_launch(void* const* d_in, const int* in_sizes, int n_in,
                              void* d_out, int out_size, void* d_ws, size_t ws_size,
                              hipStream_t stream)
{
  const float* query = (const float*)d_in[0];
  const float* key_  = (const float*)d_in[1];
  const float* value = (const float*)d_in[2];
  const int*   mask  = (const int*)d_in[3];
  const float* Wq = (const float*)d_in[4];
  const float* bq = (const float*)d_in[5];
  const float* Wk = (const float*)d_in[6];
  const float* bk = (const float*)d_in[7];
  const float* Wv = (const float*)d_in[8];
  const float* bv = (const float*)d_in[9];
  const float* Wo = (const float*)d_in[10];
  const float* bo = (const float*)d_in[11];

  char* ws = (char*)d_ws;
  _Float16* Qh  = (_Float16*)(ws);                             // 8 MB
  _Float16* Kh  = (_Float16*)(ws + (size_t)8  * 1024 * 1024);  // 8 MB
  _Float16* Vt  = (_Float16*)(ws + (size_t)16 * 1024 * 1024);  // 8 MB
  float*    ctx = (float*)   (ws + (size_t)24 * 1024 * 1024);  // 16 MB

  dim3 gg(TOKENS / 128, D_MODEL / 128);
  proj_gemm<0><<<gg, 256, 0, stream>>>(query, Wq, bq, (void*)Qh);
  proj_gemm<0><<<gg, 256, 0, stream>>>(key_,  Wk, bk, (void*)Kh);
  proj_gemm<1><<<gg, 256, 0, stream>>>(value, Wv, bv, (void*)Vt);
  flash_attn<<<dim3(BATCH * HEADS * (SEQ / 128)), 256, 0, stream>>>(Qh, Kh, Vt, mask, ctx);
  proj_gemm<2><<<gg, 256, 0, stream>>>(ctx, Wo, bo, d_out);
}
